// MultiTaskFEGIN_18940805776024
// MI455X (gfx1250) — matrix operand order repaired and checked by reading, not compiled, not run on hardware
//
#include <hip/hip_runtime.h>
#include <hip/hip_bf16.h>
#include <math.h>

#define NN 100000
#define NE 1600000
#define NG 64
#define HH 128
#define NLAY 4
#define NCLS 16
#define NCAND 100000

typedef __bf16 bf16;
typedef __attribute__((ext_vector_type(16))) __bf16 v16bf;
typedef __attribute__((ext_vector_type(8)))  __bf16 bf16x8;
typedef __attribute__((ext_vector_type(8)))  float  v8f;

static __device__ inline v8f wmma_bf16(v16bf a, v16bf b, v8f c) {
    return __builtin_amdgcn_wmma_f32_16x16x32_bf16(false, a, false, b, (short)0, c, false, false);
}

// A-fragment (16x32 bf16) from LDS row-major tile. rowbase = &A[m][kt*32].
// lane: m = lane&15, kh = lane>>4. element e -> k = 16*kh + e.
static __device__ inline v16bf load_a_frag(const bf16* rowbase, int kh) {
    union { v16bf v; bf16x8 h[2]; } u;
    const bf16x8* p = (const bf16x8*)(rowbase + 16 * kh);
    u.h[0] = p[0];   // k offsets +0..7
    u.h[1] = p[1];   // k offsets +8..15
    return u.v;
}

// ---------------- utility kernels ----------------

__global__ void zero_f32(float* __restrict__ p, long n) {
    long i = (long)blockIdx.x * blockDim.x + threadIdx.x;
    if (i < n) p[i] = 0.f;
}

// Pack W[K][Ncols] f32 -> per-fragment bf16 B layout.
// frag = kt*(Ncols/16)+nt ; within frag: lane, e ; k = kt*32 + 16*(lane>>4) + e ; n = nt*16 + (lane&15)
__global__ void pack_w(const float* __restrict__ W, bf16* __restrict__ dst, int Ncols, int total) {
    int tid = blockIdx.x * blockDim.x + threadIdx.x;
    if (tid >= total) return;
    int e    = tid & 15;
    int lane = (tid >> 4) & 31;
    int frag = tid >> 9;
    int ntiles = Ncols >> 4;
    int nt = frag % ntiles;
    int kt = frag / ntiles;
    int k = kt * 32 + ((lane >> 4) << 4) + e;
    int n = nt * 16 + (lane & 15);
    dst[tid] = (bf16)W[(long)k * Ncols + n];
}

// ---------------- GIN pipeline ----------------

// agg[dst] += h[src] for all edges; 1 thread = (edge, 4 feats)
__global__ void scatter_add(const float* __restrict__ hin, int hs,
                            const int* __restrict__ ei, float* __restrict__ agg) {
    int gid = blockIdx.x * blockDim.x + threadIdx.x;   // NE*32 threads
    int e  = gid >> 5;
    int c4 = (gid & 31) << 2;
    if (e >= NE) return;
    int s = ei[e], d = ei[NE + e];
    const float4 v = *(const float4*)(hin + (long)s * hs + c4);
    float* o = agg + (long)d * HH + c4;
    atomicAdd(o + 0, v.x); atomicAdd(o + 1, v.y);
    atomicAdd(o + 2, v.z); atomicAdd(o + 3, v.w);
}

// z = (1+eps)*h + agg ; z2 = relu(relu(z@W1+b1)@W2+b2) ; accumulate column sum/sumsq.
// 16 rows per block, 8 waves = 8 column tiles of 16. z2 may alias agg.
__global__ void __launch_bounds__(256) gin_mlp(
    const float* __restrict__ hin, int hs, const float* __restrict__ agg,
    const float* __restrict__ ceps, int layer,
    const bf16* __restrict__ pW1, const float* __restrict__ b1,
    const bf16* __restrict__ pW2, const float* __restrict__ b2,
    float* __restrict__ z2, float* __restrict__ colsum, float* __restrict__ colsq)
{
    __shared__ __align__(16) bf16 As[16][136];
    __shared__ __align__(16) bf16 Us[16][136];
    __shared__ float ssum[128], ssq[128];
    const int tid = threadIdx.x, lane = tid & 31, w = tid >> 5;
    const int row0 = blockIdx.x * 16;
    const float epsv = 1.0f + ceps[layer];
    if (tid < 128) { ssum[tid] = 0.f; ssq[tid] = 0.f; }
#pragma unroll
    for (int i = 0; i < 8; ++i) {
        int idx = i * 256 + tid;
        int r = idx >> 7, c = idx & 127;
        float z = epsv * hin[(long)(row0 + r) * hs + c] + agg[(long)(row0 + r) * HH + c];
        As[r][c] = (bf16)z;
    }
    __syncthreads();

    const int m = lane & 15, kh = lane >> 4;
    const int col = w * 16 + m;   // m == lane&15 doubles as output column within tile
    v8f acc = {0.f,0.f,0.f,0.f,0.f,0.f,0.f,0.f};
    const v16bf* B1 = (const v16bf*)pW1;
#pragma unroll
    for (int kt = 0; kt < 4; ++kt) {
        v16bf a = load_a_frag(&As[m][kt * 32], kh);
        v16bf b = B1[(kt * 8 + w) * 32 + lane];
        acc = wmma_bf16(a, b, acc);
    }
    {
        float bb = b1[col];
#pragma unroll
        for (int r = 0; r < 8; ++r) {
            float v = fmaxf(acc[r] + bb, 0.f);
            Us[r + 8 * kh][col] = (bf16)v;
        }
    }
    __syncthreads();

    v8f acc2 = {0.f,0.f,0.f,0.f,0.f,0.f,0.f,0.f};
    const v16bf* B2 = (const v16bf*)pW2;
#pragma unroll
    for (int kt = 0; kt < 4; ++kt) {
        v16bf a = load_a_frag(&Us[m][kt * 32], kh);
        v16bf b = B2[(kt * 8 + w) * 32 + lane];
        acc2 = wmma_bf16(a, b, acc2);
    }
    {
        float bb = b2[col];
        float s = 0.f, sq = 0.f;
#pragma unroll
        for (int r = 0; r < 8; ++r) {
            float v = fmaxf(acc2[r] + bb, 0.f);
            z2[(long)(row0 + r + 8 * kh) * HH + col] = v;
            s += v; sq += v * v;
        }
        atomicAdd(&ssum[col], s);
        atomicAdd(&ssq[col], sq);
    }
    __syncthreads();
    if (tid < 128) {
        atomicAdd(&colsum[tid], ssum[tid]);
        atomicAdd(&colsq[tid],  ssq[tid]);
    }
}

__global__ void bn_apply(const float* __restrict__ z2, const float* __restrict__ colsum,
                         const float* __restrict__ colsq, const float* __restrict__ gamma,
                         const float* __restrict__ beta, int layer, float* __restrict__ emb)
{
    long gid = (long)blockIdx.x * blockDim.x + threadIdx.x;   // NN*128
    int c = (int)(gid & 127);
    long row = gid >> 7;
    float mu  = colsum[c] * (1.0f / NN);
    float var = colsq[c] * (1.0f / NN) - mu * mu;
    float g = gamma[layer * HH + c], b = beta[layer * HH + c];
    float v = (z2[row * HH + c] - mu) * rsqrtf(var + 1e-5f) * g + b;
    emb[row * 512 + layer * HH + c] = v;
}

// ---------------- pooling + node classifier ----------------

__global__ void pool_kernel(const float* __restrict__ emb, const int* __restrict__ batch,
                            float* __restrict__ gemb, float* __restrict__ gcnt) {
    long gid = (long)blockIdx.x * blockDim.x + threadIdx.x;   // NN*128
    long n = gid >> 7;
    int c4 = (int)(gid & 127) << 2;
    int g = batch[n];
    const float4 v = *(const float4*)(emb + n * 512 + c4);
    float* o = gemb + (long)g * 512 + c4;
    atomicAdd(o + 0, v.x); atomicAdd(o + 1, v.y);
    atomicAdd(o + 2, v.z); atomicAdd(o + 3, v.w);
    if (c4 == 0) atomicAdd(&gcnt[g], 1.f);
}

__global__ void __launch_bounds__(256) nc_kernel(
    const float* __restrict__ gemb, const float* __restrict__ gcnt,
    const float* __restrict__ W1, const float* __restrict__ b1,
    const float* __restrict__ W2, const float* __restrict__ b2,
    const float* __restrict__ W3, const float* __restrict__ b3,
    const float* __restrict__ W4, const float* __restrict__ b4,
    float* __restrict__ out)
{
    __shared__ float row[512], t1[256], t2[128], t3[128], lg[16];
    __shared__ float lse;
    int g = blockIdx.x, tid = threadIdx.x;
    float inv = 1.f / fmaxf(gcnt[g], 1.f);
    row[tid]       = gemb[g * 512 + tid] * inv;
    row[tid + 256] = gemb[g * 512 + 256 + tid] * inv;
    __syncthreads();
    { float a = b1[tid]; for (int k = 0; k < 512; ++k) a += row[k] * W1[k * 256 + tid]; t1[tid] = fmaxf(a, 0.f); }
    __syncthreads();
    if (tid < 128) { float a = b2[tid]; for (int k = 0; k < 256; ++k) a += t1[k] * W2[k * 128 + tid]; t2[tid] = fmaxf(a, 0.f); }
    __syncthreads();
    if (tid < 128) { float a = b3[tid]; for (int k = 0; k < 128; ++k) a += t2[k] * W3[k * 128 + tid]; t3[tid] = fmaxf(a, 0.f); }
    __syncthreads();
    if (tid < 16)  { float a = b4[tid]; for (int k = 0; k < 128; ++k) a += t3[k] * W4[k * 16 + tid]; lg[tid] = a; }
    __syncthreads();
    if (tid == 0) {
        float mx = lg[0];
        for (int i = 1; i < 16; ++i) mx = fmaxf(mx, lg[i]);
        float s = 0.f;
        for (int i = 0; i < 16; ++i) s += expf(lg[i] - mx);
        lse = mx + logf(s);
    }
    __syncthreads();
    if (tid < 16) out[g * 16 + tid] = lg[tid] - lse;
}

// ---------------- edge predictor (fused gather + 2 WMMA GEMMs + dot) ----------------

__global__ void __launch_bounds__(256) ep_kernel(
    const float* __restrict__ emb, const int* __restrict__ cand,
    const bf16* __restrict__ pW1, const float* __restrict__ b1,
    const bf16* __restrict__ pW2, const float* __restrict__ b2,
    const float* __restrict__ W3, const float* __restrict__ b3,
    float* __restrict__ out)
{
    __shared__ __align__(16) bf16 As[16][1032];   // 16 x 1024 (padded)
    __shared__ __align__(16) bf16 Us[16][264];    // 16 x 256 (padded)
    __shared__ float U2[16][132];                 // 16 x 128 (padded)
    __shared__ int nd[32];
    const int tid = threadIdx.x, lane = tid & 31, w = tid >> 5;
    const int row0 = blockIdx.x * 16;
    if (tid < 16) { nd[tid] = cand[row0 + tid]; nd[16 + tid] = cand[NCAND + row0 + tid]; }
    __syncthreads();
#pragma unroll
    for (int i = 0; i < 64; ++i) {
        int idx = i * 256 + tid;
        int r = idx >> 10, c = idx & 1023;
        int node = (c < 512) ? nd[r] : nd[16 + r];
        As[r][c] = (bf16)emb[(long)node * 512 + (c & 511)];
    }
    __syncthreads();

    const int m = lane & 15, kh = lane >> 4;
    v8f a0 = {0.f,0.f,0.f,0.f,0.f,0.f,0.f,0.f};
    v8f a1 = {0.f,0.f,0.f,0.f,0.f,0.f,0.f,0.f};
    const v16bf* B1 = (const v16bf*)pW1;
#pragma unroll 4
    for (int kt = 0; kt < 32; ++kt) {
        v16bf a = load_a_frag(&As[m][kt * 32], kh);
        a0 = wmma_bf16(a, B1[(kt * 16 + w) * 32 + lane], a0);
        a1 = wmma_bf16(a, B1[(kt * 16 + w + 8) * 32 + lane], a1);
    }
    {
        int col = w * 16 + m; float bb = b1[col];
#pragma unroll
        for (int r = 0; r < 8; ++r) Us[r + 8 * kh][col] = (bf16)fmaxf(a0[r] + bb, 0.f);
        col = (w + 8) * 16 + m; bb = b1[col];
#pragma unroll
        for (int r = 0; r < 8; ++r) Us[r + 8 * kh][col] = (bf16)fmaxf(a1[r] + bb, 0.f);
    }
    __syncthreads();

    v8f c2 = {0.f,0.f,0.f,0.f,0.f,0.f,0.f,0.f};
    const v16bf* B2 = (const v16bf*)pW2;
#pragma unroll
    for (int kt = 0; kt < 8; ++kt) {
        v16bf a = load_a_frag(&Us[m][kt * 32], kh);
        c2 = wmma_bf16(a, B2[(kt * 8 + w) * 32 + lane], c2);
    }
    {
        int col = w * 16 + m; float bb = b2[col];
#pragma unroll
        for (int r = 0; r < 8; ++r) U2[r + 8 * kh][col] = fmaxf(c2[r] + bb, 0.f);
    }
    __syncthreads();
    if (tid < 16) {
        float acc = b3[0];
        for (int k = 0; k < 128; ++k) acc += U2[tid][k] * W3[k];
        out[row0 + tid] = 1.f / (1.f + expf(-acc));
    }
}

// ---------------- launch ----------------

extern "C" void kernel_launch(void* const* d_in, const int* in_sizes, int n_in,
                              void* d_out, int out_size, void* d_ws, size_t ws_size,
                              hipStream_t stream) {
    const float* x      = (const float*)d_in[0];
    const int*   ei     = (const int*)d_in[1];
    const int*   batch  = (const int*)d_in[2];
    const int*   cand   = (const int*)d_in[3];
    const float* cW1    = (const float*)d_in[4];
    const float* cb1    = (const float*)d_in[5];
    const float* cW2    = (const float*)d_in[6];
    const float* cb2    = (const float*)d_in[7];
    const float* cgamma = (const float*)d_in[8];
    const float* cbeta  = (const float*)d_in[9];
    const float* ceps   = (const float*)d_in[10];
    const float* ncW1   = (const float*)d_in[11];
    const float* ncb1   = (const float*)d_in[12];
    const float* ncW2   = (const float*)d_in[13];
    const float* ncb2   = (const float*)d_in[14];
    const float* ncW3   = (const float*)d_in[15];
    const float* ncb3   = (const float*)d_in[16];
    const float* ncW4   = (const float*)d_in[17];
    const float* ncb4   = (const float*)d_in[18];
    const float* epW1   = (const float*)d_in[19];
    const float* epb1   = (const float*)d_in[20];
    const float* epW2   = (const float*)d_in[21];
    const float* epb2   = (const float*)d_in[22];
    const float* epW3   = (const float*)d_in[23];
    const float* epb3   = (const float*)d_in[24];

    char* ws = (char*)d_ws;
    float* emb    = (float*)(ws + 0);                         // NN*512 f32
    float* aggz   = (float*)(ws + 204800000L);                // NN*128 f32 (agg, reused as z2)
    float* colsum = (float*)(ws + 256000000L);                // 128
    float* colsq  = colsum + 128;                             // 128
    float* gemb   = (float*)(ws + 256001024L);                // 64*512
    float* gcnt   = gemb + 64 * 512;                          // 64
    bf16*  pcW1   = (bf16*)(ws + 256132352L);                 // 4*16384
    bf16*  pcW2   = (bf16*)(ws + 256263424L);                 // 4*16384
    bf16*  pepW1  = (bf16*)(ws + 256394496L);                 // 262144
    bf16*  pepW2  = (bf16*)(ws + 256918784L);                 // 32768

    // pack weights into WMMA B-fragment layout (bf16)
    for (int l = 0; l < NLAY; ++l) {
        pack_w<<<64, 256, 0, stream>>>(cW1 + l * 16384, pcW1 + l * 16384, 128, 16384);
        pack_w<<<64, 256, 0, stream>>>(cW2 + l * 16384, pcW2 + l * 16384, 128, 16384);
    }
    pack_w<<<1024, 256, 0, stream>>>(epW1, pepW1, 256, 262144);
    pack_w<<<128,  256, 0, stream>>>(epW2, pepW2, 128, 32768);

    zero_f32<<<(64 * 512 + 64 + 255) / 256, 256, 0, stream>>>(gemb, 64 * 512 + 64);

    for (int l = 0; l < NLAY; ++l) {
        zero_f32<<<50000, 256, 0, stream>>>(aggz, (long)NN * HH);
        zero_f32<<<1, 256, 0, stream>>>(colsum, 256);
        const float* hin = (l == 0) ? x : (emb + (l - 1) * HH);
        int hs = (l == 0) ? HH : 512;
        scatter_add<<<200000, 256, 0, stream>>>(hin, hs, ei, aggz);
        gin_mlp<<<NN / 16, 256, 0, stream>>>(hin, hs, aggz, ceps, l,
                                             pcW1 + l * 16384, cb1 + l * HH,
                                             pcW2 + l * 16384, cb2 + l * HH,
                                             aggz, colsum, colsq);
        bn_apply<<<50000, 256, 0, stream>>>(aggz, colsum, colsq, cgamma, cbeta, l, emb);
    }

    pool_kernel<<<50000, 256, 0, stream>>>(emb, batch, gemb, gcnt);
    nc_kernel<<<NG, 256, 0, stream>>>(gemb, gcnt, ncW1, ncb1, ncW2, ncb2,
                                      ncW3, ncb3, ncW4, ncb4, (float*)d_out);
    ep_kernel<<<NCAND / 16, 256, 0, stream>>>(emb, cand, pepW1, epb1, pepW2, epb2,
                                              epW3, epb3, (float*)d_out + 1024);
}
